// TreeLSTMCell_15710990369454
// MI455X (gfx1250) — matrix operand order repaired and checked by reading, not compiled, not run on hardware
//
#include <hip/hip_runtime.h>

typedef __attribute__((ext_vector_type(16))) __bf16         v16bf;
typedef __attribute__((ext_vector_type(2)))  __bf16         v2bf;
typedef __attribute__((ext_vector_type(8)))  float          v8f;
typedef __attribute__((ext_vector_type(8)))  unsigned short v8us;
typedef __attribute__((ext_vector_type(16))) unsigned short v16us;
typedef __attribute__((ext_vector_type(4)))  unsigned int   v4ui;

#define N_NODES 65536
#define KCH     8
#define HDIM    128
#define WPAD    136                    // 17x8 bf16: 16B-aligned rows, 4-bank rotation -> conflict-free b128
#define MATSZ   (HDIM * WPAD)          // 17408: one 128x128 bf16 weight matrix, col-major [col][k]
#define ATILE   (16 * WPAD)            // 2176:  one 16x128 bf16 A tile, row-major [m][k]
#define NMAT    8                      // Wi,Ui,Wo,Uo,Wu,Uu,Wf,Uf
#define LDS_USHORTS (NMAT * MATSZ + (2 + KCH) * ATILE)  // 161024 us = 322048 B <= 327680 B (320KB)

// packed pair {lo, hi} -> one dword of two bf16
static __device__ __forceinline__ unsigned int pk_bf16(float lo, float hi) {
#if __has_builtin(__builtin_amdgcn_cvt_pk_bf16_f32)
  v2bf p = __builtin_amdgcn_cvt_pk_bf16_f32(lo, hi);
  return __builtin_bit_cast(unsigned int, p);
#else
  // one v_perm_b32: {hi[31:16], lo[31:16]}  (truncating f32->bf16)
  return __builtin_amdgcn_perm(__float_as_uint(hi), __float_as_uint(lo), 0x07060302u);
#endif
}
static __device__ __forceinline__ float fsigmoid(float x) {
  return 1.0f / (1.0f + __expf(-x));
}
static __device__ __forceinline__ float ftanh(float x) {
  float e = __expf(2.0f * x);
  return (e - 1.0f) / (e + 1.0f);
}

// 16x32 bf16 A-fragment from LDS tile stored row-major [16][WPAD] (ISA 7.12.2):
//   lanes 0-15 : M=lane,    V0-3 K=k0+0..7,  V4-7 K=k0+16..23
//   lanes16-31 : M=lane-16, V0-3 K=k0+8..15, V4-7 K=k0+24..31
static __device__ __forceinline__ v16bf load_a(const unsigned short* tile,
                                               int mn, int half, int k0) {
  const unsigned short* p = tile + mn * WPAD + k0 + 8 * half;   // 16B aligned
  v8us lo = *(const v8us*)p;
  v8us hi = *(const v8us*)(p + 16);
  v16us c = __builtin_shufflevector(lo, hi, 0,1,2,3,4,5,6,7,8,9,10,11,12,13,14,15);
  return __builtin_bit_cast(v16bf, c);
}

// 32x16 bf16 B-fragment from col-major LDS weight matrix [col][WPAD]:
//   lanes 0-15 : N=lane,    V0-3 K=k0+0..7,  V4-7 K=k0+16..23
//   lanes16-31 : N=lane-16, V0-3 K=k0+8..15, V4-7 K=k0+24..31
static __device__ __forceinline__ v16bf load_b(const unsigned short* mat,
                                               int col, int half, int k0) {
  const unsigned short* p = mat + col * WPAD + k0 + 8 * half;   // 16B aligned
  v8us lo = *(const v8us*)p;
  v8us hi = *(const v8us*)(p + 16);
  v16us c = __builtin_shufflevector(lo, hi, 0,1,2,3,4,5,6,7,8,9,10,11,12,13,14,15);
  return __builtin_bit_cast(v16bf, c);
}

#define WMMA_BF16(A, B, C) \
  __builtin_amdgcn_wmma_f32_16x16x32_bf16(false, (A), false, (B), (short)0, (C), false, false)

__global__ void __launch_bounds__(256)
treelstm_wmma(const float* __restrict__ x,
              const float* __restrict__ child_h,
              const float* __restrict__ child_c,
              const float* __restrict__ Wi, const float* __restrict__ Ui,
              const float* __restrict__ Wo, const float* __restrict__ Uo,
              const float* __restrict__ Wu, const float* __restrict__ Uu,
              const float* __restrict__ Wf, const float* __restrict__ Uf,
              const float* __restrict__ bi, const float* __restrict__ bo,
              const float* __restrict__ bu, const float* __restrict__ bf,
              float* __restrict__ out) {
  extern __shared__ unsigned short smem[];
  unsigned short* wbuf  = smem;                   // 8 weight matrices, bf16, col-major
  unsigned short* xbuf  = smem + NMAT * MATSZ;    // x tile        16x128
  unsigned short* hsbuf = xbuf + ATILE;           // h_sum tile    16x128
  unsigned short* chbuf = hsbuf + ATILE;          // child tiles 8x16x128

  const int tid  = threadIdx.x;
  const int lane = tid & 31;
  const int wv   = tid >> 5;      // wave 0..7 -> output columns wv*16..wv*16+15
  const int half = lane >> 4;
  const int mn   = lane & 15;     // A-row (m) / B,C-column (n)
  const int col0 = wv * 16;
  const int sm   = tid >> 4;      // staging: node row 0..15
  const int d0   = (tid & 15) * 8;// staging: 8-element chunk of a 128-wide row

  // ---- one-time: stage the 8 weight matrices into LDS (bf16, transposed) ----
  // Each thread converts two consecutive K rows at one column -> packed b32 store.
  {
    const float* Ws[NMAT] = {Wi, Ui, Wo, Uo, Wu, Uu, Wf, Uf};
    #pragma unroll 1
    for (int m = 0; m < NMAT; ++m) {
      const float* src = Ws[m];
      unsigned short* dst = wbuf + m * MATSZ;
      #pragma unroll 1
      for (int idx = tid; idx < (HDIM / 2) * HDIM; idx += 256) {
        int kp = idx >> 7, c = idx & 127;   // coalesced row-major reads of rows 2kp, 2kp+1
        unsigned int pk = pk_bf16(src[(2 * kp) * HDIM + c], src[(2 * kp + 1) * HDIM + c]);
        *(unsigned int*)(dst + c * WPAD + 2 * kp) = pk;   // transposed, 4B aligned
      }
    }
  }

  const float bi_l = bi[col0 + mn];
  const float bo_l = bo[col0 + mn];
  const float bu_l = bu[col0 + mn];
  const float bf_l = bf[col0 + mn];

  const unsigned short* WiB = wbuf + 0 * MATSZ;
  const unsigned short* UiB = wbuf + 1 * MATSZ;
  const unsigned short* WoB = wbuf + 2 * MATSZ;
  const unsigned short* UoB = wbuf + 3 * MATSZ;
  const unsigned short* WuB = wbuf + 4 * MATSZ;
  const unsigned short* UuB = wbuf + 5 * MATSZ;
  const unsigned short* WfB = wbuf + 6 * MATSZ;
  const unsigned short* UfB = wbuf + 7 * MATSZ;

  const int numTiles = N_NODES / 16;
  for (int tile = blockIdx.x; tile < numTiles; tile += gridDim.x) {
    const int node0 = tile * 16;
    __syncthreads();  // weights staged / previous tile fully consumed

    // ---- stage x, child_h (bf16) and h_sum into LDS (b128 in, b128 out) ----
    {
      const float4* xp = (const float4*)(x + (size_t)(node0 + sm) * HDIM + d0);
      float4 a0 = xp[0], a1 = xp[1];
      v4ui px = {pk_bf16(a0.x, a0.y), pk_bf16(a0.z, a0.w),
                 pk_bf16(a1.x, a1.y), pk_bf16(a1.z, a1.w)};
      *(v4ui*)(xbuf + sm * WPAD + d0) = px;

      float hs0 = 0, hs1 = 0, hs2 = 0, hs3 = 0, hs4 = 0, hs5 = 0, hs6 = 0, hs7 = 0;
      #pragma unroll
      for (int k = 0; k < KCH; ++k) {
        const float4* cp =
            (const float4*)(child_h + ((size_t)(node0 + sm) * KCH + k) * HDIM + d0);
        float4 c0 = cp[0], c1 = cp[1];
        v4ui pc = {pk_bf16(c0.x, c0.y), pk_bf16(c0.z, c0.w),
                   pk_bf16(c1.x, c1.y), pk_bf16(c1.z, c1.w)};
        *(v4ui*)(chbuf + k * ATILE + sm * WPAD + d0) = pc;
        hs0 += c0.x; hs1 += c0.y; hs2 += c0.z; hs3 += c0.w;
        hs4 += c1.x; hs5 += c1.y; hs6 += c1.z; hs7 += c1.w;
      }
      v4ui ph = {pk_bf16(hs0, hs1), pk_bf16(hs2, hs3),
                 pk_bf16(hs4, hs5), pk_bf16(hs6, hs7)};
      *(v4ui*)(hsbuf + sm * WPAD + d0) = ph;
    }
    __syncthreads();

    // ---- prefetch next tile's streaming inputs while we compute this one ----
    {
      const int ntile = tile + (int)gridDim.x;
      if (ntile < numTiles) {
        const int nn0 = ntile * 16;
        __builtin_prefetch(x + (size_t)(nn0 + sm) * HDIM + d0, 0, 0);
        #pragma unroll
        for (int k = 0; k < KCH; ++k)
          __builtin_prefetch(child_h + ((size_t)(nn0 + sm) * KCH + k) * HDIM + d0, 0, 0);
      }
    }

    // ---- gate GEMMs over K=128 in 4 steps of 32 ----
    v8f accI = {}, accO = {}, accU = {}, accF = {};
    v16bf bUf[4];                         // Uf B-fragments, reused by all 8 children
    #pragma unroll
    for (int kt = 0; kt < 4; ++kt) {
      const int k0 = kt * 32;
      v16bf aX = load_a(xbuf, mn, half, k0);
      v16bf aH = load_a(hsbuf, mn, half, k0);
      bUf[kt] = load_b(UfB, col0 + mn, half, k0);
      accI = WMMA_BF16(aX, load_b(WiB, col0 + mn, half, k0), accI);
      accI = WMMA_BF16(aH, load_b(UiB, col0 + mn, half, k0), accI);
      accO = WMMA_BF16(aX, load_b(WoB, col0 + mn, half, k0), accO);
      accO = WMMA_BF16(aH, load_b(UoB, col0 + mn, half, k0), accO);
      accU = WMMA_BF16(aX, load_b(WuB, col0 + mn, half, k0), accU);
      accU = WMMA_BF16(aH, load_b(UuB, col0 + mn, half, k0), accU);
      accF = WMMA_BF16(aX, load_b(WfB, col0 + mn, half, k0), accF);
    }

    // ---- per-child forget gates: f_k = sigmoid(fx + b_f + child_h_k @ Uf) ----
    v8f accC = {};
    #pragma unroll
    for (int k = 0; k < KCH; ++k) {
      v8f g = accF;
      #pragma unroll
      for (int kt = 0; kt < 4; ++kt) {
        v16bf aC = load_a(chbuf + k * ATILE, mn, half, kt * 32);
        g = WMMA_BF16(aC, bUf[kt], g);
      }
      const float* ccp = child_c + ((size_t)node0 * KCH + k) * HDIM + col0 + mn;
      #pragma unroll
      for (int r = 0; r < 8; ++r) {
        int m = r + 8 * half;                        // C-frag row mapping
        float f  = fsigmoid(g[r] + bf_l);
        float cc = ccp[(size_t)m * KCH * HDIM];      // child_c[node0+m][k][col]
        accC[r] += f * cc;
      }
    }

    // ---- epilogue: c = i*u + sum_k f_k*c_k ; h = o*tanh(c) ----
    float* hout = out + (size_t)node0 * HDIM + col0 + mn;
    float* cout = out + (size_t)N_NODES * HDIM + (size_t)node0 * HDIM + col0 + mn;
    #pragma unroll
    for (int r = 0; r < 8; ++r) {
      int m = r + 8 * half;
      float iv = fsigmoid(accI[r] + bi_l);
      float ov = fsigmoid(accO[r] + bo_l);
      float uv = ftanh(accU[r] + bu_l);
      float cv = iv * uv + accC[r];
      float hv = ov * ftanh(cv);
      hout[(size_t)m * HDIM] = hv;
      cout[(size_t)m * HDIM] = cv;
    }
  }
}

extern "C" void kernel_launch(void* const* d_in, const int* in_sizes, int n_in,
                              void* d_out, int out_size, void* d_ws, size_t ws_size,
                              hipStream_t stream) {
  (void)in_sizes; (void)n_in; (void)out_size; (void)d_ws; (void)ws_size;
  const float* x       = (const float*)d_in[0];
  const float* child_h = (const float*)d_in[1];
  const float* child_c = (const float*)d_in[2];
  const float* Wi = (const float*)d_in[3];
  const float* bi = (const float*)d_in[4];
  const float* Ui = (const float*)d_in[5];
  const float* Wo = (const float*)d_in[6];
  const float* bo = (const float*)d_in[7];
  const float* Uo = (const float*)d_in[8];
  const float* Wu = (const float*)d_in[9];
  const float* bu = (const float*)d_in[10];
  const float* Uu = (const float*)d_in[11];
  const float* Wf = (const float*)d_in[12];
  const float* bf = (const float*)d_in[13];
  const float* Uf = (const float*)d_in[14];

  size_t shmem = (size_t)LDS_USHORTS * sizeof(unsigned short);  // 322048 B
  (void)hipFuncSetAttribute((const void*)treelstm_wmma,
                            hipFuncAttributeMaxDynamicSharedMemorySize, (int)shmem);
  treelstm_wmma<<<dim3(512), dim3(256), shmem, stream>>>(
      x, child_h, child_c, Wi, Ui, Wo, Uo, Wu, Uu, Wf, Uf,
      bi, bo, bu, bf, (float*)d_out);
}
